// RNNBase_37203006718229
// MI455X (gfx1250) — matrix operand order repaired and checked by reading, not compiled, not run on hardware
//
#include <hip/hip_runtime.h>
#include <math.h>

// GRU: T=512, B=64, H=1024, L=2. PyTorch gate order r|z|n.
#define T_  512
#define B_  64
#define H_  1024
#define L_  2
#define N3H 3072   // 3*H
#define K_  1024   // input/hidden width per layer

typedef __attribute__((ext_vector_type(16))) __bf16 v16bf;
typedef __attribute__((ext_vector_type(8)))  float  v8f;

__device__ __forceinline__ unsigned short f2bf(float f) {
    unsigned u = __float_as_uint(f);
    return (unsigned short)((u + 0x7FFFu + ((u >> 16) & 1u)) >> 16);  // RNE
}
__device__ __forceinline__ float bf2f(unsigned short h) {
    return __uint_as_float(((unsigned)h) << 16);
}

union BfFrag {
    v16bf v;
    unsigned short u[16];
    uint4 q[2];
};

// Async DMA of 16 bytes global -> LDS (no VGPR bounce), tracked by ASYNCcnt.
// LDS address = low 32 bits of the generic pointer (flat->LDS truncation).
__device__ __forceinline__ void async_b128(void* ldsDst, const void* gSrc) {
    unsigned lds = (unsigned)(unsigned long long)ldsDst;
    asm volatile("global_load_async_to_lds_b128 %0, %1, off"
                 :: "v"(lds), "v"(gSrc) : "memory");
}
__device__ __forceinline__ void wait_async0() {
    asm volatile("s_wait_asynccnt 0x0" ::: "memory");
}

// ---------------------------------------------------------------- prep kernels
__global__ void cvt_f32_bf16(const float* __restrict__ in,
                             unsigned short* __restrict__ out, int n) {
    int i = blockIdx.x * blockDim.x + threadIdx.x;
    if (i < n) out[i] = f2bf(in[i]);
}

__global__ void copy_f32(const float* __restrict__ in,
                         float* __restrict__ out, int n) {
    int i = blockIdx.x * blockDim.x + threadIdx.x;
    if (i < n) out[i] = in[i];
}

// ---------------------------------------------------------------- WMMA GEMM
// out[M,3H] = A[M,K] (bf16) @ W[3H,K]^T (bf16) + bias, f32 accumulate.
// Workgroup tile TM x 64, K-chunks of 32 double-buffered through LDS with
// async-to-LDS DMA overlapping the WMMA compute of the previous chunk.
//   TM=128 -> 8 waves, each 16x64 (4 acc tiles, 1 shared A frag)
//   TM=64  -> 8 waves, each 16x32 (2 acc tiles, 1 shared A frag)
template <int TM, bool OUT_BF16>
__global__ __launch_bounds__(256)
void gemm_bf16_lds(const unsigned short* __restrict__ A,   // [M, K_] bf16 row-major
                   const unsigned short* __restrict__ W,   // [N3H, K_] bf16 (B^T)
                   const float* __restrict__ bias,         // [N3H]
                   float* __restrict__ outF,               // if !OUT_BF16
                   unsigned short* __restrict__ outB,      // if OUT_BF16
                   int M) {
    constexpr int LP  = 40;            // padded LDS row pitch (ushorts, 80B)
    constexpr int TPW = TM / 32;       // accumulator tiles per wave (4 or 2)
    constexpr int NB  = N3H / 64;      // 48 column blocks
    constexpr int ACH = (TM * 4) / 256;// A chunks per thread per stage (2 or 1)

    __shared__ unsigned short sA[2][TM * LP];
    __shared__ unsigned short sB[2][64 * LP];

    const int bm = blockIdx.x / NB;
    const int bn = blockIdx.x % NB;
    const int tid  = threadIdx.x;
    const int wave = tid >> 5;
    const int lane = tid & 31;
    const int laneLo = lane & 15;
    const int hi     = lane >> 4;

    // wave -> (mi, niBase): tiles enumerated (mi = t/4, ni = t%4), TPW per wave
    const int t0 = wave * TPW;
    const int mi = t0 >> 2;
    const int niBase = t0 & 3;

    // accumulators init = bias broadcast per column
    v8f acc[TPW];
#pragma unroll
    for (int j = 0; j < TPW; ++j) {
        float bv = bias[bn * 64 + (niBase + j) * 16 + laneLo];
#pragma unroll
        for (int i = 0; i < 8; ++i) acc[j][i] = bv;
    }

    const unsigned short* Ablk = A + (size_t)(bm * TM) * K_;
    const unsigned short* Wblk = W + (size_t)(bn * 64) * K_;

    // Branch-free async stage of one K-chunk (A: TM x 32, B: 64 x 32) into buf.
    auto stage = [&](int k, int buf) {
#pragma unroll
        for (int j = 0; j < ACH; ++j) {
            int c = tid + j * 256;
            int row = c >> 2, kc = (c & 3) * 8;
            async_b128(&sA[buf][row * LP + kc], &Ablk[(size_t)row * K_ + k + kc]);
        }
        {
            int row = tid >> 2, kc = (tid & 3) * 8;   // 256 chunks == 1 per thread
            async_b128(&sB[buf][row * LP + kc], &Wblk[(size_t)row * K_ + k + kc]);
        }
    };

    stage(0, 0);
    for (int k = 0; k < K_; k += 32) {
        const int cur = (k >> 5) & 1;
        wait_async0();        // my stage into buf[cur] is complete
        __syncthreads();      // everyone's stage is complete
        if (k + 32 < K_)      // uniform branch: kick next stage into other buffer
            stage(k + 32, cur ^ 1);

        // A fragment (16x32): lanes 0-15 rows, K{0..7,16..23}; lanes 16-31 K{8..15,24..31}
        BfFrag a;
        {
            int row = mi * 16 + laneLo;
            int ka = hi * 8;
            a.q[0] = *(const uint4*)&sA[cur][row * LP + ka];
            a.q[1] = *(const uint4*)&sA[cur][row * LP + ka + 16];
        }
        // B fragments (32x16): lane = column, lanes 0-15 K{0..7,16..23}; lanes 16-31 K{8..15,24..31}
#pragma unroll
        for (int j = 0; j < TPW; ++j) {
            BfFrag b;
            int col = (niBase + j) * 16 + laneLo;
            int kb = hi * 8;
            b.q[0] = *(const uint4*)&sB[cur][col * LP + kb];
            b.q[1] = *(const uint4*)&sB[cur][col * LP + kb + 16];
            acc[j] = __builtin_amdgcn_wmma_f32_16x16x32_bf16(
                false, a.v, false, b.v, (short)0, acc[j], false, false);
        }
        __syncthreads();      // all reads of buf[cur] done before it is re-staged
    }

    // store: D VGPR r -> row mi*16 + r + hi*8, column = tile base + laneLo
#pragma unroll
    for (int j = 0; j < TPW; ++j) {
#pragma unroll
        for (int r = 0; r < 8; ++r) {
            size_t idx = (size_t)(bm * TM + mi * 16 + r + hi * 8) * N3H
                       + (bn * 64 + (niBase + j) * 16 + laneLo);
            if (OUT_BF16) outB[idx] = f2bf(acc[j][r]);
            else          outF[idx] = acc[j][r];
        }
    }
    (void)M;
}

// ---------------------------------------------------------------- gate fusion
// r = sig(xr+hr); z = sig(xz+hz); n = tanh(xn + r*hn); h' = (1-z)n + z h
template <bool SEQ_BF16>
__global__ __launch_bounds__(256)
void gru_elem(int t,
              const unsigned short* __restrict__ xg,   // [T,B,3H] bf16 (bias incl.)
              const float* __restrict__ hg,            // [B,3H]   f32  (bias incl.)
              float* __restrict__ h,                   // [B,H] f32, in place
              unsigned short* __restrict__ h_bf,       // [B,H] bf16
              unsigned short* __restrict__ seq_bf,     // [T,B,H] bf16 (layer 0)
              float* __restrict__ seq_f) {             // [T,B,H] f32  (layer 1)
    int idx = blockIdx.x * blockDim.x + threadIdx.x;
    if (idx >= B_ * H_) return;
    int b = idx >> 10;           // / H_
    int c = idx & (H_ - 1);      // % H_
    size_t xbase = ((size_t)t * B_ + b) * N3H;
    size_t hbase = (size_t)b * N3H;
    float xr = bf2f(xg[xbase + c]);
    float xz = bf2f(xg[xbase + H_ + c]);
    float xn = bf2f(xg[xbase + 2 * H_ + c]);
    float hr = hg[hbase + c];
    float hz = hg[hbase + H_ + c];
    float hn = hg[hbase + 2 * H_ + c];
    float r = 1.f / (1.f + __expf(-(xr + hr)));
    float z = 1.f / (1.f + __expf(-(xz + hz)));
    float n = tanhf(xn + r * hn);
    float hnew = (1.f - z) * n + z * h[idx];
    h[idx] = hnew;
    h_bf[idx] = f2bf(hnew);
    size_t sidx = (size_t)t * (B_ * H_) + idx;
    if (SEQ_BF16) seq_bf[sidx] = f2bf(hnew);
    else          seq_f[sidx] = hnew;
}

// ---------------------------------------------------------------- launch
extern "C" void kernel_launch(void* const* d_in, const int* in_sizes, int n_in,
                              void* d_out, int out_size, void* d_ws, size_t ws_size,
                              hipStream_t stream) {
    const float* x   = (const float*)d_in[0];  // [T,B,H]
    const float* h0  = (const float*)d_in[1];  // [L,B,H]
    const float* Wih = (const float*)d_in[2];  // [L,3H,H]
    const float* Whh = (const float*)d_in[3];  // [L,3H,H]
    const float* bih = (const float*)d_in[4];  // [L,3H]
    const float* bhh = (const float*)d_in[5];  // [L,3H]
    float* out = (float*)d_out;                // [T,B,H] ++ h_n [L,B,H]

    // Workspace carve-up (~345 MB)
    char*  ws  = (char*)d_ws;
    size_t off = 0;
    auto carve = [&](size_t bytes) -> char* {
        char* p = ws + off;
        off += (bytes + 255) & ~(size_t)255;
        return p;
    };
    const size_t nWl = (size_t)N3H * K_;  // weight elems per layer
    unsigned short* WihB = (unsigned short*)carve(L_ * nWl * 2);                 // 12 MB
    unsigned short* WhhB = (unsigned short*)carve(L_ * nWl * 2);                 // 12 MB
    unsigned short* xgB  = (unsigned short*)carve((size_t)T_ * B_ * N3H * 2);    // 192 MB
    unsigned short* xbf  = (unsigned short*)carve((size_t)T_ * B_ * H_ * 2);     // 64 MB
    unsigned short* x1bf = (unsigned short*)carve((size_t)T_ * B_ * H_ * 2);     // 64 MB
    unsigned short* hbf  = (unsigned short*)carve((size_t)B_ * H_ * 2);          // 128 KB
    float*          hg   = (float*)carve((size_t)B_ * N3H * 4);                  // 768 KB
    (void)ws_size; (void)in_sizes; (void)n_in; (void)out_size;

    // One-time conversions; h_n region of d_out doubles as live fp32 hidden state.
    const int nW = (int)(L_ * nWl);
    cvt_f32_bf16<<<(nW + 255) / 256, 256, 0, stream>>>(Wih, WihB, nW);
    cvt_f32_bf16<<<(nW + 255) / 256, 256, 0, stream>>>(Whh, WhhB, nW);
    const int nX = T_ * B_ * H_;
    cvt_f32_bf16<<<(nX + 255) / 256, 256, 0, stream>>>(x, xbf, nX);
    const int nH0 = L_ * B_ * H_;
    copy_f32<<<(nH0 + 255) / 256, 256, 0, stream>>>(h0, out + (size_t)T_ * B_ * H_, nH0);

    const int NB = N3H / 64;  // 48 column blocks
    for (int l = 0; l < L_; ++l) {
        const unsigned short* Ain = (l == 0) ? xbf : x1bf;
        // Input-gate GEMM: xg = Ain @ Wih^T + bih  (M=32768, N=3072, K=1024)
        {
            int M = T_ * B_;
            gemm_bf16_lds<128, true><<<(M / 128) * NB, 256, 0, stream>>>(
                Ain, WihB + (size_t)l * nWl, bih + (size_t)l * N3H,
                nullptr, xgB, M);
        }
        float* h = out + (size_t)T_ * B_ * H_ + (size_t)l * B_ * H_;
        // seed bf16 copy of h0[l]
        cvt_f32_bf16<<<(B_ * H_) / 256, 256, 0, stream>>>(h, hbf, B_ * H_);

        for (int tt = 0; tt < T_; ++tt) {
            // hg = h @ Whh^T + bhh (M=64, N=3072, K=1024); weights stay L2-resident
            gemm_bf16_lds<64, false><<<NB, 256, 0, stream>>>(
                hbf, WhhB + (size_t)l * nWl, bhh + (size_t)l * N3H,
                hg, nullptr, B_);
            if (l == 0) {
                gru_elem<true><<<(B_ * H_) / 256, 256, 0, stream>>>(
                    tt, xgB, hg, h, hbf, x1bf, nullptr);
            } else {
                gru_elem<false><<<(B_ * H_) / 256, 256, 0, stream>>>(
                    tt, xgB, hg, h, hbf, nullptr, out);
            }
        }
    }
}
